// IntrapersonGraphLayer_39556648796606
// MI455X (gfx1250) — matrix operand order repaired and checked by reading, not compiled, not run on hardware
//
#include <hip/hip_runtime.h>

// ---------------- problem constants ----------------
#define N_   16384
#define V_   17
#define D_   128
#define H_   256
#define E_   32
#define BT   8          // batch elements per workgroup (M=16 rows = 8 batch x 2 edges/joints)

typedef __attribute__((ext_vector_type(16))) _Float16       v16h;
typedef __attribute__((ext_vector_type(8)))  _Float16       v8h;
typedef __attribute__((ext_vector_type(16))) unsigned short v16us;
typedef __attribute__((ext_vector_type(8)))  float          v8f;

// ---------------- dynamic LDS layout (bytes) ----------------
#define OFF_HLDS   0u        // f16 [BT*17*128]               34816
#define OFF_AGG    34816u    // f32 [BT*17*128]               69632
#define OFF_WS1    104448u   // f16 [256][256] transposed     131072 (ew1/nw1)
#define OFF_WS2    235520u   // f16 [128][256] transposed     65536  (ew2/nw2)
#define OFF_HID    301056u   // union: f16[16*256] hid | f32[16*128] xbuf  8192
#define OFF_RELW   309248u   // f32 [2][256] (ew1 rows 256,257) 2048
#define OFF_REL    311296u   // f32 [BT*32][2]                2048
#define OFF_VALID  313344u   // f32 [BT*32]                   1024
#define OFF_DENOM  314368u   // f32 [BT*17]                   544
#define OFF_EB1    314912u   // f32 [256]
#define OFF_EB2    315936u   // f32 [128]
#define OFF_NB1    316448u   // f32 [256]
#define OFF_NB2    317472u   // f32 [128]
#define OFF_GAM    317984u   // f32 [128]
#define OFF_BET    318496u   // f32 [128]
#define OFF_MASKF  319008u   // f32 [BT*17]                   544
#define OFF_DST    319552u   // i32 [32]
#define OFF_SRC    319680u   // i32 [32]
#define OFF_MUS    319808u   // f32 [16]
#define OFF_RSTD   319872u   // f32 [16]
#define SMEM_BYTES 319936u

// Build A fragment (16x32, not the ISA k order): lane M = lane&15,
// 16 contiguous K at [kb + koff .. +15], koff = (lane>>4)*16.
__device__ __forceinline__ v16h load_a_f16(const _Float16* p) {
    v8h lo = *(const v8h*)p;
    v8h hi = *(const v8h*)(p + 8);
    return __builtin_shufflevector(lo, hi, 0,1,2,3,4,5,6,7,8,9,10,11,12,13,14,15);
}
__device__ __forceinline__ v16h load_a_f32(const float* p) {
    v16h a;
#pragma unroll
    for (int j = 0; j < 8; ++j) { a[j] = (_Float16)p[j]; a[j + 8] = (_Float16)p[8 + j]; }
    return a;
}
// B fragment (32x16, not the ISA k order): lane N = lane&15, 16 contiguous K at (lane>>4)*16.
__device__ __forceinline__ v16h load_b(const _Float16* q) {
    v16us raw = *(const v16us*)q;                 // 32B aligned LDS load
    return __builtin_bit_cast(v16h, raw);
}

__global__ __launch_bounds__(256, 1)
void poseGnnFused(const float* __restrict__ hG,  const float* __restrict__ xyG,
                  const unsigned char* __restrict__ maskG, const int* __restrict__ eiG,
                  const float* __restrict__ ew1G, const float* __restrict__ eb1G,
                  const float* __restrict__ ew2G, const float* __restrict__ eb2G,
                  const float* __restrict__ nw1G, const float* __restrict__ nb1G,
                  const float* __restrict__ nw2G, const float* __restrict__ nb2G,
                  const float* __restrict__ gamG, const float* __restrict__ betG,
                  float* __restrict__ outG)
{
    extern __shared__ unsigned char smem[];
    _Float16* hlds  = (_Float16*)(smem + OFF_HLDS);
    float*    agg   = (float*)   (smem + OFF_AGG);
    _Float16* ws1   = (_Float16*)(smem + OFF_WS1);
    _Float16* ws2   = (_Float16*)(smem + OFF_WS2);
    _Float16* hid   = (_Float16*)(smem + OFF_HID);
    float*    xbuf  = (float*)   (smem + OFF_HID);   // union with hid
    float*    relw  = (float*)   (smem + OFF_RELW);
    float*    rel   = (float*)   (smem + OFF_REL);
    float*    valid = (float*)   (smem + OFF_VALID);
    float*    denom = (float*)   (smem + OFF_DENOM);
    float*    seb1  = (float*)   (smem + OFF_EB1);
    float*    seb2  = (float*)   (smem + OFF_EB2);
    float*    snb1  = (float*)   (smem + OFF_NB1);
    float*    snb2  = (float*)   (smem + OFF_NB2);
    float*    sgam  = (float*)   (smem + OFF_GAM);
    float*    sbet  = (float*)   (smem + OFF_BET);
    float*    maskf = (float*)   (smem + OFF_MASKF);
    int*      dste  = (int*)     (smem + OFF_DST);
    int*      srce  = (int*)     (smem + OFF_SRC);
    float*    mus   = (float*)   (smem + OFF_MUS);
    float*    rstds = (float*)   (smem + OFF_RSTD);

    const int tid  = threadIdx.x;
    const int wv   = tid >> 5;
    const int lane = tid & 31;
    const int lhi  = lane >> 4;       // 0/1 (lane group)
    const int llo  = lane & 15;
    const int koff = lhi * 16;
    const int n0   = blockIdx.x * BT;

    // Warm L2/WGP$ for the node-phase weights (used only after the edge phase):
    // global_prefetch_b8, one 128B line per lane per iteration.
#pragma unroll
    for (int i = 0; i < 8; ++i)
        __builtin_prefetch((const char*)nw1G + (i * 256 + tid) * 128, 0, 1);
#pragma unroll
    for (int i = 0; i < 4; ++i)
        __builtin_prefetch((const char*)nw2G + (i * 256 + tid) * 128, 0, 1);

    // ---------------- phase 0: stage h tile, tables, edge weights ----------------
    {
        const float4* h4 = (const float4*)(hG + (size_t)n0 * V_ * D_);
        for (int i = tid; i < BT * V_ * D_ / 4; i += 256) {
            float4 f = h4[i];
            hlds[4*i+0] = (_Float16)f.x; hlds[4*i+1] = (_Float16)f.y;
            hlds[4*i+2] = (_Float16)f.z; hlds[4*i+3] = (_Float16)f.w;
        }
        for (int i = tid; i < BT * V_ * D_; i += 256) agg[i] = 0.f;
        if (tid < E_) { dste[tid] = eiG[2*tid]; srce[tid] = eiG[2*tid+1]; }
        if (tid < BT * V_) {
            int m = tid / V_, v = tid % V_;
            maskf[tid] = maskG[(size_t)(n0 + m) * V_ + v] ? 1.f : 0.f;
        }
        for (int i = tid; i < 512; i += 256) relw[i] = ew1G[(256 + (i >> 8)) * H_ + (i & 255)];
        if (tid < H_) { seb1[tid] = eb1G[tid]; snb1[tid] = nb1G[tid]; }
        if (tid < D_) {
            seb2[tid] = eb2G[tid]; snb2[tid] = nb2G[tid];
            sgam[tid] = gamG[tid]; sbet[tid] = betG[tid];
        }
        // ew1 rows 0..255 -> ws1[c][k], ew2 -> ws2[c][k] (transposed, f16)
        for (int i = tid; i < 256 * H_; i += 256) { int k = i >> 8, c = i & 255; ws1[c*256 + k] = (_Float16)ew1G[(size_t)k*H_ + c]; }
        for (int i = tid; i < H_ * D_;  i += 256) { int k = i >> 7, c = i & 127; ws2[c*256 + k] = (_Float16)ew2G[(size_t)k*D_ + c]; }
    }
    __syncthreads();

    // rel & edge_valid : exactly 256 = BT*E threads
    {
        int m = tid >> 5, e = tid & 31;
        int dv = dste[e], sv = srce[e];
        const float* xyB = xyG + (size_t)(n0 + m) * V_ * 2;
        rel[tid*2+0] = xyB[sv*2+0] - xyB[dv*2+0];
        rel[tid*2+1] = xyB[sv*2+1] - xyB[dv*2+1];
        valid[tid]   = maskf[m*V_ + dv] * maskf[m*V_ + sv];
    }
    __syncthreads();
    if (tid < BT * V_) {
        int m = tid / V_, v = tid % V_;
        float s = 0.f;
        for (int e = 0; e < E_; ++e) if (dste[e] == v) s += valid[m*E_ + e];
        denom[tid] = s;
    }
    __syncthreads();

    // ---------------- edge MLP: 16 groups of 2 edges (M = 8 batch x 2 edges) ----------------
    for (int eg = 0; eg < 16; ++eg) {
        const int cb0 = wv * 32, cb1 = cb0 + 16;
        v8f c0, c1;
        { float b0 = seb1[cb0 + llo], b1 = seb1[cb1 + llo];
#pragma unroll
          for (int i = 0; i < 8; ++i) { c0[i] = b0; c1[i] = b1; } }
        const int Ma = llo, ma = Ma & 7, ea = 2*eg + (Ma >> 3);
        const int dva = dste[ea], sva = srce[ea];
#pragma unroll
        for (int kt = 0; kt < 8; ++kt) {            // K = 256 (target|source)
            const int vrow = (kt < 4) ? dva : sva;
            v16h a  = load_a_f16(hlds + (ma*V_ + vrow)*D_ + (kt & 3)*32 + koff);
            v16h b0 = load_b(ws1 + (cb0 + llo)*256 + kt*32 + lhi*16);
            v16h b1 = load_b(ws1 + (cb1 + llo)*256 + kt*32 + lhi*16);
            c0 = __builtin_amdgcn_wmma_f32_16x16x32_f16(false, a, false, b0, (short)0, c0, false, false);
            c1 = __builtin_amdgcn_wmma_f32_16x16x32_f16(false, a, false, b1, (short)0, c1, false, false);
        }
        // rank-2 rel update (ew1 rows 256,257) + bias already in C + ReLU -> hid (f16)
#pragma unroll
        for (int i = 0; i < 8; ++i) {
            int M = i + 8*lhi, m = M & 7, e = 2*eg + (M >> 3);
            float rx = rel[(m*E_ + e)*2 + 0], ry = rel[(m*E_ + e)*2 + 1];
            int ca = cb0 + llo, cb = cb1 + llo;
            float h0 = c0[i] + rx*relw[ca] + ry*relw[256 + ca];
            float h1 = c1[i] + rx*relw[cb] + ry*relw[256 + cb];
            hid[M*256 + ca] = (_Float16)fmaxf(h0, 0.f);
            hid[M*256 + cb] = (_Float16)fmaxf(h1, 0.f);
        }
        __syncthreads();
        // second layer: msg = relu(...) @ ew2 + eb2, then masked scatter-add into agg
        const int c2 = wv*16 + llo;
        v8f dacc;
        { float b2 = seb2[c2];
#pragma unroll
          for (int i = 0; i < 8; ++i) dacc[i] = b2; }
#pragma unroll
        for (int kt = 0; kt < 8; ++kt) {
            v16h a = load_a_f16(hid + llo*256 + kt*32 + koff);
            v16h b = load_b(ws2 + c2*256 + kt*32 + lhi*16);
            dacc = __builtin_amdgcn_wmma_f32_16x16x32_f16(false, a, false, b, (short)0, dacc, false, false);
        }
#pragma unroll
        for (int i = 0; i < 8; ++i) {
            int M = i + 8*lhi, m = M & 7, e = 2*eg + (M >> 3);
            float val = dacc[i] * valid[m*E_ + e];
            atomicAdd(&agg[(m*V_ + dste[e])*D_ + c2], val);   // ds_add_f32
        }
        __syncthreads();
    }

    // normalize agg by max(denom,1)
    for (int i = tid; i < BT * V_ * D_; i += 256)
        agg[i] = agg[i] / fmaxf(denom[i >> 7], 1.f);
    __syncthreads();

    // restage node weights into the same LDS buffers (transposed, f16)
    for (int i = tid; i < 256 * H_; i += 256) { int k = i >> 8, c = i & 255; ws1[c*256 + k] = (_Float16)nw1G[(size_t)k*H_ + c]; }
    for (int i = tid; i < H_ * D_;  i += 256) { int k = i >> 7, c = i & 127; ws2[c*256 + k] = (_Float16)nw2G[(size_t)k*D_ + c]; }
    __syncthreads();

    // ---------------- node MLP + residual + LayerNorm: 9 groups of 2 joints ----------------
    for (int vg = 0; vg < 9; ++vg) {
        const int v0 = 2*vg, v1 = (2*vg + 1 < V_) ? 2*vg + 1 : V_ - 1;
        const int cb0 = wv * 32, cb1 = cb0 + 16;
        v8f c0, c1;
        { float b0 = snb1[cb0 + llo], b1 = snb1[cb1 + llo];
#pragma unroll
          for (int i = 0; i < 8; ++i) { c0[i] = b0; c1[i] = b1; } }
        const int Ma = llo, ma = Ma & 7, va = (Ma >> 3) ? v1 : v0;
#pragma unroll
        for (int kt = 0; kt < 8; ++kt) {            // K = 256 (h | agg)
            v16h a = (kt < 4)
                ? load_a_f16(hlds + (ma*V_ + va)*D_ + kt*32 + koff)
                : load_a_f32(agg  + (ma*V_ + va)*D_ + (kt - 4)*32 + koff);
            v16h b0 = load_b(ws1 + (cb0 + llo)*256 + kt*32 + lhi*16);
            v16h b1 = load_b(ws1 + (cb1 + llo)*256 + kt*32 + lhi*16);
            c0 = __builtin_amdgcn_wmma_f32_16x16x32_f16(false, a, false, b0, (short)0, c0, false, false);
            c1 = __builtin_amdgcn_wmma_f32_16x16x32_f16(false, a, false, b1, (short)0, c1, false, false);
        }
#pragma unroll
        for (int i = 0; i < 8; ++i) {
            int M = i + 8*lhi;
            hid[M*256 + cb0 + llo] = (_Float16)fmaxf(c0[i], 0.f);
            hid[M*256 + cb1 + llo] = (_Float16)fmaxf(c1[i], 0.f);
        }
        __syncthreads();
        const int c2 = wv*16 + llo;
        v8f dacc;
        { float b2 = snb2[c2];
#pragma unroll
          for (int i = 0; i < 8; ++i) dacc[i] = b2; }
#pragma unroll
        for (int kt = 0; kt < 8; ++kt) {
            v16h a = load_a_f16(hid + llo*256 + kt*32 + koff);
            v16h b = load_b(ws2 + c2*256 + kt*32 + lhi*16);
            dacc = __builtin_amdgcn_wmma_f32_16x16x32_f16(false, a, false, b, (short)0, dacc, false, false);
        }
        __syncthreads();                             // hid reads done before xbuf (union) write
#pragma unroll
        for (int i = 0; i < 8; ++i) {
            int M = i + 8*lhi, m = M & 7, v = (M >> 3) ? v1 : v0;
            float hval = (float)hlds[(m*V_ + v)*D_ + c2];
            float hn   = denom[m*V_ + v] > 0.f ? 1.f : 0.f;
            xbuf[M*D_ + c2] = hval + dacc[i] * hn;   // x = h + delta*has_neighbors
        }
        __syncthreads();
        // LayerNorm stats: 8 threads per row, 16 contiguous f32 each, wave32 butterfly
        if (tid < 128) {
            int M = tid >> 3, cb = (tid & 7) * 16;
            float s = 0.f, ss = 0.f;
#pragma unroll
            for (int c = 0; c < 16; ++c) { float xv = xbuf[M*D_ + cb + c]; s += xv; ss += xv*xv; }
#pragma unroll
            for (int o = 1; o < 8; o <<= 1) {
                s  += __shfl_xor(s,  o, 32);
                ss += __shfl_xor(ss, o, 32);
            }
            if ((tid & 7) == 0) {
                float mu  = s * (1.f / D_);
                float var = ss * (1.f / D_) - mu * mu;
                mus[M]   = mu;
                rstds[M] = __frsqrt_rn(var + 1e-5f);
            }
        }
        __syncthreads();
        for (int i = tid; i < 16 * D_; i += 256) {
            int M = i >> 7, c = i & 127;
            if (!(vg == 8 && M >= 8)) {              // skip duplicated pad row
                int m = M & 7, v = (M >> 3) ? v1 : v0;
                float o = (xbuf[i] - mus[M]) * rstds[M] * sgam[c] + sbet[c];
                outG[(((size_t)(n0 + m)) * V_ + v) * D_ + c] = o * maskf[m*V_ + v];
            }
        }
        __syncthreads();
    }
}

extern "C" void kernel_launch(void* const* d_in, const int* in_sizes, int n_in,
                              void* d_out, int out_size, void* d_ws, size_t ws_size,
                              hipStream_t stream) {
    (void)in_sizes; (void)n_in; (void)out_size; (void)d_ws; (void)ws_size;
    const float* h   = (const float*)d_in[0];
    const float* xy  = (const float*)d_in[1];
    const unsigned char* jm = (const unsigned char*)d_in[2];
    const int*   ei  = (const int*)d_in[3];
    const float* ew1 = (const float*)d_in[4];
    const float* eb1 = (const float*)d_in[5];
    const float* ew2 = (const float*)d_in[6];
    const float* eb2 = (const float*)d_in[7];
    const float* nw1 = (const float*)d_in[8];
    const float* nb1 = (const float*)d_in[9];
    const float* nw2 = (const float*)d_in[10];
    const float* nb2 = (const float*)d_in[11];
    const float* gm  = (const float*)d_in[12];
    const float* bt  = (const float*)d_in[13];
    dim3 grid(N_ / BT), block(256);
    hipLaunchKernelGGL(poseGnnFused, grid, block, SMEM_BYTES, stream,
                       h, xy, jm, ei, ew1, eb1, ew2, eb2, nw1, nb1, nw2, nb2, gm, bt,
                       (float*)d_out);
}
